// Attention_63952063037719
// MI455X (gfx1250) — matrix operand order repaired and checked by reading, not compiled, not run on hardware
//
#include <hip/hip_runtime.h>
#include <hip/hip_bf16.h>
#include <stdint.h>

// ---------------------------------------------------------------------------
// Attention: out = softmax((xWq^T)(xWk^T)^T / sqrt(D)) (xWv^T), heads=16
// B=64, N=577, E=1024, H=16, D=64.  All GEMMs via v_wmma_f32_16x16x32_bf16.
// Shared tiles (x rows in the projection, K/V tiles in attention) are staged
// through LDS with global_load_async_to_lds_b128 (ASYNCcnt double buffering).
// ---------------------------------------------------------------------------

typedef __attribute__((ext_vector_type(16))) __bf16 v16bf;
typedef __attribute__((ext_vector_type(8)))  __bf16 v8bf;
typedef __attribute__((ext_vector_type(4)))  __bf16 v4bf;
typedef __attribute__((ext_vector_type(8)))  float  v8f;

#define BATCH 64
#define SEQ   577
#define EMB   1024
#define NHEAD 16
#define HDIM  64
#define ROWS  (BATCH * SEQ)    // 36928 = 1154 * 32
#define NPAD  608              // SEQ padded to multiple of 32 (19 key blocks)
#define CHUNK 128              // K-chunk staged in LDS per double-buffer slot
#define MTILE 32               // rows per block in the projection GEMM

static __device__ __forceinline__ v8f wmma_bf16(v16bf a, v16bf b, v8f c) {
  return __builtin_amdgcn_wmma_f32_16x16x32_bf16(false, a, false, b,
                                                 (short)0, c, false, false);
}

// 16-bit A-fragment: lane holds row M=lane%16; elems 0..7 = K[kOff..],
// elems 8..15 = K[kOff+8..], kOff = (lane/16)*16 (contiguous k; B frags match).
static __device__ __forceinline__ v16bf load_afrag(const __bf16* p0,
                                                   const __bf16* p1) {
  v8bf lo = *(const v8bf*)p0;
  v8bf hi = *(const v8bf*)p1;
  v16bf a;
#pragma unroll
  for (int i = 0; i < 8; ++i) { a[i] = lo[i]; a[i + 8] = hi[i]; }
  return a;
}

// Async global -> LDS 128-bit copy (ISA 15.18.3 op 98, tracked by ASYNCcnt).
static __device__ __forceinline__ void async_ld_b128(uint32_t lds_off,
                                                     const __bf16* gptr) {
  asm volatile("global_load_async_to_lds_b128 %0, %1, off"
               :
               : "v"(lds_off), "v"(gptr)
               : "memory");
}

// ------------------------- stage 0: fp32 -> bf16 ---------------------------
__global__ void cvt_f32_bf16(const float* __restrict__ src,
                             __bf16* __restrict__ dst, int n4) {
  int i = blockIdx.x * blockDim.x + threadIdx.x;
  if (i < n4) {
    float4 v = ((const float4*)src)[i];
    v4bf o;
    o[0] = (__bf16)v.x; o[1] = (__bf16)v.y;
    o[2] = (__bf16)v.z; o[3] = (__bf16)v.w;
    ((v4bf*)dst)[i] = o;
  }
}

__global__ void zero_b128(uint4* __restrict__ dst, int n) {
  int i = blockIdx.x * blockDim.x + threadIdx.x;
  if (i < n) dst[i] = uint4{0u, 0u, 0u, 0u};
}

// ------------------ stage 1: QKV projection GEMM (bf16 WMMA) ---------------
// grid = (1154 row-tiles of 32, 4 col-groups of 256, 3 matrices), block 128.
// Each wave: 32x64 output tile = 8 accumulators; per k32 step it reads two
// A-frags from the LDS-staged x tile and 4 W B-frags from global -> 8 WMMAs
// per 128 B/lane of B traffic (32 FLOP/B from L2).
__global__ __launch_bounds__(128) void qkv_gemm(
    const __bf16* __restrict__ xb, const __bf16* __restrict__ wb,
    __bf16* __restrict__ qb, __bf16* __restrict__ kbuf,
    __bf16* __restrict__ vtb) {
  __shared__ __align__(16) __bf16 Abuf[2][MTILE][CHUNK];  // 2 x 8 KB

  const int t    = threadIdx.x;
  const int lane = t & 31;
  const int wave = t >> 5;
  const int lr   = lane & 15;
  const int lh   = lane >> 4;
  const int mt    = blockIdx.x;
  const int obase = blockIdx.y * 256 + wave * 64;
  const int z     = blockIdx.z;                 // 0=Q 1=K 2=V
  const __bf16* W = wb + (size_t)z * EMB * EMB; // row o of W == B-frag col o

  // 512 16-byte staging slots per chunk; thread t owns slots t+i*128.
  const uint32_t lbase = (uint32_t)(uintptr_t)(&Abuf[0][0][0]);
  const __bf16* g[4];
  uint32_t l[4];
#pragma unroll
  for (int i = 0; i < 4; ++i) {
    const int s = t + i * 128;
    g[i] = xb + (size_t)(mt * MTILE + (s >> 4)) * EMB + (s & 15) * 8;
    l[i] = lbase + (uint32_t)s * 16;
  }

  // prologue: chunk 0 -> buffer 0
#pragma unroll
  for (int i = 0; i < 4; ++i) async_ld_b128(l[i], g[i]);

  v8f acc[8] = {};
  for (int c = 0; c < EMB / CHUNK; ++c) {   // 8 chunks
    const int kk = c * CHUNK;
    if (c + 1 < EMB / CHUNK) {
      const uint32_t nb = (uint32_t)((c + 1) & 1) * (MTILE * CHUNK * 2);
#pragma unroll
      for (int i = 0; i < 4; ++i) async_ld_b128(l[i] + nb, g[i] + kk + CHUNK);
      asm volatile("s_wait_asynccnt 0x4" ::: "memory");  // chunk c landed
    } else {
      asm volatile("s_wait_asynccnt 0x0" ::: "memory");
    }
    __syncthreads();  // every wave waited its own ASYNCcnt -> tile visible

    const __bf16* Ach = &Abuf[c & 1][0][0];
#pragma unroll
    for (int ks = 0; ks < CHUNK; ks += 32) {
      const __bf16* ar0 = Ach + lr * CHUNK + ks;          // rows 0..15
      const __bf16* ar1 = Ach + (16 + lr) * CHUNK + ks;   // rows 16..31
      v16bf a0 = load_afrag(ar0 + lh * 16, ar0 + lh * 16 + 8);
      v16bf a1 = load_afrag(ar1 + lh * 16, ar1 + lh * 16 + 8);
      const __bf16* wk = W + kk + ks + lh * 16;
      v16bf b0 = *(const v16bf*)(wk + (size_t)(obase + 0  + lr) * EMB);
      v16bf b1 = *(const v16bf*)(wk + (size_t)(obase + 16 + lr) * EMB);
      v16bf b2 = *(const v16bf*)(wk + (size_t)(obase + 32 + lr) * EMB);
      v16bf b3 = *(const v16bf*)(wk + (size_t)(obase + 48 + lr) * EMB);
      acc[0] = wmma_bf16(a0, b0, acc[0]);
      acc[1] = wmma_bf16(a0, b1, acc[1]);
      acc[2] = wmma_bf16(a0, b2, acc[2]);
      acc[3] = wmma_bf16(a0, b3, acc[3]);
      acc[4] = wmma_bf16(a1, b0, acc[4]);
      acc[5] = wmma_bf16(a1, b1, acc[5]);
      acc[6] = wmma_bf16(a1, b2, acc[6]);
      acc[7] = wmma_bf16(a1, b3, acc[7]);
    }
    __syncthreads();  // reads done before the buffer is overwritten
  }

#pragma unroll
  for (int gq = 0; gq < 2; ++gq)
#pragma unroll
    for (int f = 0; f < 4; ++f) {
      const int o = obase + f * 16 + lr;       // output feature column
      const int h = o >> 6, d = o & 63;
#pragma unroll
      for (int j = 0; j < 8; ++j) {
        const int r  = mt * MTILE + gq * 16 + j + lh * 8;  // flattened (b,n)
        const int bi = r / SEQ;
        const int n  = r - bi * SEQ;
        const __bf16 val = (__bf16)acc[gq * 4 + f][j];
        const size_t bh = (size_t)bi * NHEAD + h;
        if (z == 0)      qb  [(bh * NPAD + n) * HDIM + d] = val;
        else if (z == 1) kbuf[(bh * NPAD + n) * HDIM + d] = val;
        else             vtb [(bh * HDIM + d) * NPAD + n] = val;
      }
    }
}

// --------------- stage 2: flash attention (online softmax) -----------------
// Block = 4 waves on the same (b,h), adjacent 16-query tiles; the 32-key
// K-tile (32x64) and V-tile (64x32) are shared: staged via async LDS loads,
// double-buffered.  8 WMMAs per wave per key-block.
__global__ __launch_bounds__(128) void attn_fwd(
    const __bf16* __restrict__ qb, const __bf16* __restrict__ kbuf,
    const __bf16* __restrict__ vtb, float* __restrict__ out) {
  __shared__ __align__(16) __bf16 Kt[2][32][HDIM];  // 2 x 4 KB
  __shared__ __align__(16) __bf16 Vt2[2][HDIM][32]; // 2 x 4 KB
  __shared__ __align__(16) __bf16 plds[4][16][32];  // per-wave P tile

  const int tid  = threadIdx.x;
  const int lane = tid & 31;
  const int wave = tid >> 5;
  const int lr   = lane & 15;
  const int lh   = lane >> 4;

  const int bh = blockIdx.x / 10;          // b*16 + h
  const int qg = blockIdx.x % 10;
  int qt = qg * 4 + wave;
  if (qt > 36) qt = 36;                    // clamp: duplicate work, same data
  const int h = bh & 15, bi = bh >> 4;
  const int qbase = qt * 16;

  const __bf16* Q  = qb   + (size_t)bh * NPAD * HDIM;
  const __bf16* K  = kbuf + (size_t)bh * NPAD * HDIM;
  const __bf16* Vt = vtb  + (size_t)bh * HDIM * NPAD;

  // staging slots: K tile 256x16B (row s/8, 16B granule s%8),
  //                V tile 256x16B (row s/4, granule s%4); 2 of each per thread
  const uint32_t lK = (uint32_t)(uintptr_t)(&Kt[0][0][0]);
  const uint32_t lV = (uint32_t)(uintptr_t)(&Vt2[0][0][0]);
  const int sk0 = tid, sk1 = tid + 128;
  const size_t kOff0 = (size_t)(sk0 >> 3) * HDIM + (sk0 & 7) * 8;
  const size_t kOff1 = (size_t)(sk1 >> 3) * HDIM + (sk1 & 7) * 8;
  const size_t vOff0 = (size_t)(sk0 >> 2) * NPAD + (sk0 & 3) * 8;
  const size_t vOff1 = (size_t)(sk1 >> 2) * NPAD + (sk1 & 3) * 8;

  // Q fragments for d=0..31 and d=32..63
  const __bf16* qr = Q + (size_t)(qbase + lr) * HDIM;
  const v16bf aq0 = load_afrag(qr + lh * 16,      qr + lh * 16 + 8);
  const v16bf aq1 = load_afrag(qr + 32 + lh * 16, qr + 40 + lh * 16);

  float mr[8], lsum[8];
  v8f oacc[4] = {};
#pragma unroll
  for (int j = 0; j < 8; ++j) { mr[j] = -1e30f; lsum[j] = 0.f; }

  // prologue: key-block 0 -> buffer 0
  async_ld_b128(lK + (uint32_t)sk0 * 16, K + kOff0);
  async_ld_b128(lK + (uint32_t)sk1 * 16, K + kOff1);
  async_ld_b128(lV + (uint32_t)sk0 * 16, Vt + vOff0);
  async_ld_b128(lV + (uint32_t)sk1 * 16, Vt + vOff1);

  for (int it = 0; it < NPAD / 32; ++it) {   // 19 key blocks
    const int kb = it * 32;
    if (it + 1 < NPAD / 32) {
      const uint32_t nb = (uint32_t)((it + 1) & 1) * 4096;
      const size_t gk = (size_t)(kb + 32) * HDIM;
      async_ld_b128(lK + nb + (uint32_t)sk0 * 16, K + gk + kOff0);
      async_ld_b128(lK + nb + (uint32_t)sk1 * 16, K + gk + kOff1);
      async_ld_b128(lV + nb + (uint32_t)sk0 * 16, Vt + (kb + 32) + vOff0);
      async_ld_b128(lV + nb + (uint32_t)sk1 * 16, Vt + (kb + 32) + vOff1);
      asm volatile("s_wait_asynccnt 0x4" ::: "memory");
    } else {
      asm volatile("s_wait_asynccnt 0x0" ::: "memory");
    }
    __syncthreads();

    const int cur = it & 1;
    // --- scores S = Q K^T (two 16-col tiles), K B-frags from LDS ---
    const __bf16* k0 = &Kt[cur][lr][lh * 16];
    const __bf16* k1 = &Kt[cur][16 + lr][lh * 16];
    v16bf bk0a = *(const v16bf*)(k0);
    v16bf bk0b = *(const v16bf*)(k0 + 32);
    v16bf bk1a = *(const v16bf*)(k1);
    v16bf bk1b = *(const v16bf*)(k1 + 32);
    v8f zc = {};
    v8f s0 = wmma_bf16(aq0, bk0a, zc); s0 = wmma_bf16(aq1, bk0b, s0);
    v8f s1 = wmma_bf16(aq0, bk1a, zc); s1 = wmma_bf16(aq1, bk1b, s1);

    const bool v0 = (kb + lr)      < SEQ;   // column validity (padding mask)
    const bool v1 = (kb + 16 + lr) < SEQ;
    float ss0[8], ss1[8], tmx[8];
#pragma unroll
    for (int j = 0; j < 8; ++j) {
      ss0[j] = v0 ? s0[j] * 0.125f : -1e30f;   // 1/sqrt(64)
      ss1[j] = v1 ? s1[j] * 0.125f : -1e30f;
      tmx[j] = fmaxf(ss0[j], ss1[j]);
    }
    // row max: butterfly inside each 16-lane half (rows live across 16 lanes)
#pragma unroll
    for (int msk = 1; msk <= 8; msk <<= 1)
#pragma unroll
      for (int j = 0; j < 8; ++j)
        tmx[j] = fmaxf(tmx[j], __shfl_xor(tmx[j], msk, 32));

    float p0[8], p1[8], rs[8];
#pragma unroll
    for (int j = 0; j < 8; ++j) {
      const float mn = fmaxf(mr[j], tmx[j]);
      const float sc = __expf(mr[j] - mn);
      p0[j] = __expf(ss0[j] - mn);
      p1[j] = __expf(ss1[j] - mn);
      mr[j] = mn;
      lsum[j] *= sc;
      oacc[0][j] *= sc; oacc[1][j] *= sc; oacc[2][j] *= sc; oacc[3][j] *= sc;
      rs[j] = p0[j] + p1[j];
    }
#pragma unroll
    for (int msk = 1; msk <= 8; msk <<= 1)
#pragma unroll
      for (int j = 0; j < 8; ++j) rs[j] += __shfl_xor(rs[j], msk, 32);
#pragma unroll
    for (int j = 0; j < 8; ++j) lsum[j] += rs[j];

    // --- C-layout P -> A-layout via per-wave LDS tile ---
#pragma unroll
    for (int j = 0; j < 8; ++j) {
      const int row = j + lh * 8;
      plds[wave][row][lr]      = (__bf16)p0[j];
      plds[wave][row][16 + lr] = (__bf16)p1[j];
    }
    __syncthreads();
    const __bf16* pr = &plds[wave][lr][0];
    v16bf ap = load_afrag(pr + lh * 16, pr + lh * 16 + 8);

    // --- O += P V : V B-frags from LDS (row = head dim, keys contiguous) ---
    const __bf16* vbp = &Vt2[cur][lr][lh * 16];
    v16bf bv0 = *(const v16bf*)(vbp);
    v16bf bv1 = *(const v16bf*)(vbp + 16 * 32);
    v16bf bv2 = *(const v16bf*)(vbp + 32 * 32);
    v16bf bv3 = *(const v16bf*)(vbp + 48 * 32);
    oacc[0] = wmma_bf16(ap, bv0, oacc[0]);
    oacc[1] = wmma_bf16(ap, bv1, oacc[1]);
    oacc[2] = wmma_bf16(ap, bv2, oacc[2]);
    oacc[3] = wmma_bf16(ap, bv3, oacc[3]);
    __syncthreads();  // all reads done before next chunk overwrites buffers
  }

  // epilogue: normalize, merge heads, store fp32
#pragma unroll
  for (int f = 0; f < 4; ++f)
#pragma unroll
    for (int j = 0; j < 8; ++j) {
      const int n = qbase + j + lh * 8;
      if (n < SEQ)
        out[((size_t)bi * SEQ + n) * EMB + h * HDIM + f * 16 + lr] =
            oacc[f][j] / lsum[j];
    }
}

// ---------------------------------------------------------------------------
extern "C" void kernel_launch(void* const* d_in, const int* in_sizes, int n_in,
                              void* d_out, int out_size, void* d_ws,
                              size_t ws_size, hipStream_t stream) {
  (void)in_sizes; (void)n_in; (void)out_size; (void)ws_size;
  const float* x  = (const float*)d_in[0];
  const float* Wq = (const float*)d_in[1];
  const float* Wk = (const float*)d_in[2];
  const float* Wv = (const float*)d_in[3];

  char* ws = (char*)d_ws;
  const size_t xb_bytes  = (size_t)ROWS * EMB * 2;                 // 75.6 MB
  const size_t wb_bytes  = 3ull * EMB * EMB * 2;                   //  6.3 MB
  const size_t qkv_elems = (size_t)BATCH * NHEAD * NPAD * HDIM;    // 39.8M
  const size_t qkv_bytes = qkv_elems * 2;                          // 79.7 MB

  __bf16* xb   = (__bf16*)(ws);
  __bf16* wb   = (__bf16*)(ws + xb_bytes);
  __bf16* qb   = (__bf16*)(ws + xb_bytes + wb_bytes);
  __bf16* kbuf = (__bf16*)((char*)qb + qkv_bytes);
  __bf16* vtb  = (__bf16*)((char*)kbuf + qkv_bytes);

  // stage 0: bf16 conversion
  cvt_f32_bf16<<<(ROWS * EMB / 4 + 255) / 256, 256, 0, stream>>>(x, xb,
                                                                ROWS * EMB / 4);
  const int w4 = EMB * EMB / 4;
  cvt_f32_bf16<<<(w4 + 255) / 256, 256, 0, stream>>>(Wq, wb, w4);
  cvt_f32_bf16<<<(w4 + 255) / 256, 256, 0, stream>>>(Wk, wb + (size_t)EMB * EMB, w4);
  cvt_f32_bf16<<<(w4 + 255) / 256, 256, 0, stream>>>(Wv, wb + 2ull * EMB * EMB, w4);

  // zero-fill padded Q/K/Vt region (softmax masking relies on zero pads)
  const int nz = (int)(3 * qkv_bytes / 16);
  zero_b128<<<(nz + 255) / 256, 256, 0, stream>>>((uint4*)qb, nz);

  // stage 1: QKV projections
  dim3 g1(ROWS / MTILE, 4, 3);
  qkv_gemm<<<g1, 128, 0, stream>>>(xb, wb, qb, kbuf, vtb);

  // stage 2: attention: (b,h) x 10 query-groups, 4 waves per block
  attn_fwd<<<BATCH * NHEAD * 10, 128, 0, stream>>>(qb, kbuf, vtb,
                                                   (float*)d_out);
}
